// Linear_Fitting_21328807592203
// MI455X (gfx1250) — matrix operand order repaired and checked by reading, not compiled, not run on hardware
//
#include <hip/hip_runtime.h>

// ---------------------------------------------------------------------------
// Problem constants (match reference)
// ---------------------------------------------------------------------------
#define B_  4
#define S_  1024
#define D_  1024
#define NH  16
#define HD  64
#define O_  1024   // == S_

typedef __bf16 bf16;
typedef __attribute__((ext_vector_type(16))) __bf16 v16bf;
typedef __attribute__((ext_vector_type(8)))  __bf16 v8bf;
typedef __attribute__((ext_vector_type(8)))  float  v8f;
typedef __attribute__((ext_vector_type(4)))  unsigned int u32x4;
typedef __attribute__((ext_vector_type(8)))  int    i32x8;
typedef __attribute__((ext_vector_type(4)))  int    i32x4;

#if __has_builtin(__builtin_amdgcn_tensor_load_to_lds) && \
    __has_builtin(__builtin_amdgcn_s_wait_tensorcnt)
#define USE_TDM 1
#else
#define USE_TDM 0
#endif

// ---------------------------------------------------------------------------
// WMMA helper — v_wmma_f32_16x16x32_bf16, D = A(16x32) * B(32x16) + C
// ---------------------------------------------------------------------------
__device__ __forceinline__ v8f wmma_bf16(v16bf a, v16bf b, v8f c) {
  return __builtin_amdgcn_wmma_f32_16x16x32_bf16(false, a, false, b, (short)0, c,
                                                 false, false);
}

// ---------------------------------------------------------------------------
// Tensor Data Mover: 2D tile (bf16) global -> LDS.  D# packed per CDNA5 ISA
// §8.3/§8.4 (group0: count|lds_addr|global_addr|type=2; group1: data_size=2B,
// tensor dims == tile dims (no OOB), dim0 stride in elements).
// This toolchain exposes the 6-arg builtin:
//   (u32x4 g0, i32x8 g1, i32x4 g2, i32x4 g3, i32x8 ext, i32 cpol)
// ---------------------------------------------------------------------------
#if USE_TDM
__device__ __forceinline__ void tdm_load_2d(unsigned lds_addr, const void* gsrc,
                                            unsigned tile_d0, unsigned tile_d1,
                                            unsigned long long stride_elems) {
  const unsigned long long ga = (unsigned long long)(size_t)gsrc;
  u32x4 g0 = {
      1u,                                            // count=1, user mode
      lds_addr,                                      // LDS byte address
      (unsigned)(ga & 0xFFFFFFFFu),                  // global_addr[31:0]
      (unsigned)((ga >> 32) & 0x01FFFFFFu) | (2u << 30)  // addr[56:32] | type=2
  };
  i32x8 g1 = {
      (int)(1u << 16),                               // mask=0, data_size=1 (2B)
      (int)((tile_d0 & 0xFFFFu) << 16),              // tensor_dim0[15:0]
      (int)((tile_d0 >> 16) | ((tile_d1 & 0xFFFFu) << 16)),  // d0[31:16], d1[15:0]
      (int)((tile_d1 >> 16) | (tile_d0 << 16)),      // d1[31:16], tile_dim0
      (int)(tile_d1 & 0xFFFFu),                      // tile_dim1, tile_dim2=0
      (int)(stride_elems & 0xFFFFFFFFull),           // dim0_stride[31:0]
      (int)((stride_elems >> 32) & 0xFFFFull),       // dim0_stride[47:32]
      0
  };
  i32x4 z4 = {0, 0, 0, 0};
  i32x8 z8 = {0, 0, 0, 0, 0, 0, 0, 0};
  __builtin_amdgcn_tensor_load_to_lds(g0, g1, z4, z4, z8, 0);
}
#endif

// Cooperative fallback staging: vectorized 16B copies, all threads participate.
__device__ __forceinline__ void stage_tile_coop(bf16* __restrict__ dst,
                                                const bf16* __restrict__ src,
                                                size_t rstride, int rows, int cols,
                                                int tid, int nthr) {
  const int c8 = cols >> 3;
  for (int i = tid; i < rows * c8; i += nthr) {
    const int r = i / c8, c = (i % c8) << 3;
    ((v8bf*)dst)[i] = *(const v8bf*)(src + (size_t)r * rstride + c);
  }
}

// Unified tile loader: TDM (wave 0 issues DMA) or cooperative fallback.
__device__ __forceinline__ void load_tile(bf16* dst, const bf16* src, size_t rstride,
                                          int rows, int cols, int tid, int nthr,
                                          int wave) {
#if USE_TDM
  (void)tid; (void)nthr;
  if (wave == 0)
    tdm_load_2d((unsigned)(size_t)dst, src, (unsigned)cols, (unsigned)rows, rstride);
#else
  (void)wave;
  stage_tile_coop(dst, src, rstride, rows, cols, tid, nthr);
#endif
}

// Wait for staged tiles (TENSORcnt on the issuing wave) + workgroup barrier.
__device__ __forceinline__ void tiles_commit(int wave) {
#if USE_TDM
  if (wave == 0) __builtin_amdgcn_s_wait_tensorcnt(0);
#else
  (void)wave;
#endif
  __syncthreads();
}

// ---------------------------------------------------------------------------
// Fragment loaders from LDS tiles (row width 64 bf16 = 128B rows)
// A layout: lane m=lane%16; lanes<16 K{0..15}+k0, lanes>=16 K{16..31}+k0
// ---------------------------------------------------------------------------
__device__ __forceinline__ v16bf lds_a64(const bf16 (*T)[64], int lane, int r0,
                                         int k0) {
  const int m  = lane & 15;
  const int kb = (lane < 16) ? 0 : 16;
  const bf16* p = &T[r0 + m][k0];
  v16bf a;
#pragma unroll
  for (int v = 0; v < 8; ++v) {
    const int kk = kb + (v << 1);
    a[2 * v]     = p[kk];
    a[2 * v + 1] = p[kk + 1];
  }
  return a;
}

// B from an [N][K]-layout tile (row n, contiguous k): lane n=lane%16,
// lanes<16 K 0..15, lanes>=16 K 16..31 (each lane: 16 contiguous bf16)
__device__ __forceinline__ v16bf lds_bT64(const bf16 (*T)[64], int lane, int n0,
                                          int k0) {
  const int n  = lane & 15;
  const int kb = (lane < 16) ? 0 : 16;
  const bf16* p = &T[n0 + n][k0 + kb];
  v16bf b;
#pragma unroll
  for (int v = 0; v < 8; ++v) {
    b[2 * v]     = p[2 * v];
    b[2 * v + 1] = p[2 * v + 1];
  }
  return b;
}

// B from a [K][N]-layout tile (row k, contiguous n)
__device__ __forceinline__ v16bf lds_bK64(const bf16 (*T)[64], int lane, int k0,
                                          int n0) {
  const int n  = lane & 15;
  const int kb = (lane < 16) ? 0 : 16;
  v16bf b;
#pragma unroll
  for (int v = 0; v < 8; ++v) {
    b[2 * v]     = T[k0 + kb + 2 * v][n0 + n];
    b[2 * v + 1] = T[k0 + kb + 2 * v + 1][n0 + n];
  }
  return b;
}

// ---------------------------------------------------------------------------
// K0: f32 -> bf16 conversion
// ---------------------------------------------------------------------------
__global__ void k_cvt(const float* __restrict__ in, bf16* __restrict__ out, int n) {
  int i = blockIdx.x * 256 + threadIdx.x;
  if (i < n) out[i] = (bf16)in[i];
}

// ---------------------------------------------------------------------------
// K1: P = softmax64( A @ W^T + bias ).  A:[M,K], W:[N,K] bf16 row-major.
// Block 256 thr = 8 waves (2Mx4N) -> 32x64 tile == one softmax chunk.
// K-loop: TDM stages A(32x64) + W(64x64) tiles into LDS, 2 WMMAs per wave.
// ---------------------------------------------------------------------------
__global__ void k_gemm_softmax(const bf16* __restrict__ A, const bf16* __restrict__ W,
                               const float* __restrict__ bias, bf16* __restrict__ P,
                               int M, int K, int N) {
  __shared__ alignas(16) bf16 As[32][64];
  __shared__ alignas(16) bf16 Bs[64][64];
  __shared__ float hs[32][64];
  const int tid  = threadIdx.x;
  const int lane = tid & 31, wave = tid >> 5;
  const int wm = wave >> 2, wn = wave & 3;
  const size_t rowblk = (size_t)blockIdx.x * 32;
  const size_t colblk = (size_t)blockIdx.y * 64;

  v8f acc = {};
  for (int k0 = 0; k0 < K; k0 += 64) {
    __syncthreads();  // previous tiles fully consumed
    load_tile(&As[0][0], A + rowblk * (size_t)K + k0, K, 32, 64, tid, 256, wave);
    load_tile(&Bs[0][0], W + colblk * (size_t)K + k0, K, 64, 64, tid, 256, wave);
    tiles_commit(wave);
    v16bf a = lds_a64(As, lane, wm * 16, 0);
    v16bf b = lds_bT64(Bs, lane, wn * 16, 0);
    acc = wmma_bf16(a, b, acc);
    a = lds_a64(As, lane, wm * 16, 32);
    b = lds_bT64(Bs, lane, wn * 16, 32);
    acc = wmma_bf16(a, b, acc);
  }

  // C layout: col = lane%16, rows r + (lane<16?0:8)
  {
    const int cc = lane & 15;
    const int rb = (lane < 16) ? 0 : 8;
    const float bb = bias[colblk + wn * 16 + cc];
#pragma unroll
    for (int r = 0; r < 8; ++r)
      hs[wm * 16 + rb + r][wn * 16 + cc] = acc[r] + bb;
  }
  __syncthreads();

  // fused softmax over the 64-wide head chunk
  const int row = tid >> 3, seg = tid & 7;
  float mx = -3.4e38f;
#pragma unroll
  for (int j = 0; j < 8; ++j) mx = fmaxf(mx, hs[row][seg * 8 + j]);
#pragma unroll
  for (int off = 4; off >= 1; off >>= 1) mx = fmaxf(mx, __shfl_xor(mx, off, 8));
  float e[8];
  float sum = 0.f;
#pragma unroll
  for (int j = 0; j < 8; ++j) {
    e[j] = __expf(hs[row][seg * 8 + j] - mx);
    sum += e[j];
  }
#pragma unroll
  for (int off = 4; off >= 1; off >>= 1) sum += __shfl_xor(sum, off, 8);
  const float inv = 1.0f / sum;

  bf16* dst = P + ((size_t)blockIdx.x * 32 + row) * (size_t)N + colblk + seg * 8;
#pragma unroll
  for (int j = 0; j < 8; ++j) dst[j] = (bf16)(e[j] * inv);
}

// ---------------------------------------------------------------------------
// K2: xsum[b,n,h] = sum_s p[b,n,s,h]
// ---------------------------------------------------------------------------
__global__ void k_xsum(const bf16* __restrict__ P, float* __restrict__ xsum) {
  __shared__ float red[4][HD];
  const int bn = blockIdx.x;
  const int b = bn / NH, n = bn % NH;
  const int h = threadIdx.x & 63, c = threadIdx.x >> 6;
  const bf16* p = P + (size_t)b * S_ * D_ + (size_t)n * HD + h;
  float s = 0.f;
  for (int ss = c; ss < S_; ss += 4) s += (float)p[(size_t)ss * D_];
  red[c][h] = s;
  __syncthreads();
  if (c == 0)
    xsum[(size_t)bn * HD + h] = red[0][h] + red[1][h] + red[2][h] + red[3][h];
}

// ---------------------------------------------------------------------------
// K3: t[b,n,e,o] = sum_h attn_w[o,n,e,h] * xsum[b,n,h]
// attn_weight (256 MiB) read exactly once; one wave per o, reused over 4 b.
// ---------------------------------------------------------------------------
__global__ void k_contract(const float* __restrict__ W, const float* __restrict__ xs,
                           bf16* __restrict__ T) {
  const int lane = threadIdx.x & 31, wave = threadIdx.x >> 5;
  const int o = blockIdx.x * 8 + wave;
  const int e = blockIdx.y, n = blockIdx.z;
  const float* wp = W + (((size_t)o * NH + n) * HD + e) * HD;
  const float w0 = wp[lane], w1 = wp[lane + 32];
  float partial[B_];
#pragma unroll
  for (int b = 0; b < B_; ++b) {
    const float* x = xs + ((size_t)b * NH + n) * HD;
    partial[b] = w0 * x[lane] + w1 * x[lane + 32];
  }
#pragma unroll
  for (int off = 16; off >= 1; off >>= 1) {
#pragma unroll
    for (int b = 0; b < B_; ++b) partial[b] += __shfl_xor(partial[b], off, 32);
  }
  if (lane == 0) {
#pragma unroll
    for (int b = 0; b < B_; ++b)
      T[(((size_t)b * NH + n) * HD + e) * O_ + o] = (bf16)partial[b];
  }
}

// ---------------------------------------------------------------------------
// K4: flash attention per (b,n): Q=p [S,64], K^T=t [64,O], V=p [O,64].
// Never materializes the 256 MiB score tensor.  Block = 128 thr (4 waves)
// per 16 query rows; K^T and V 64x64 tiles staged into LDS by the TDM.
// ---------------------------------------------------------------------------
__global__ void k_flash(const bf16* __restrict__ P, const bf16* __restrict__ T,
                        bf16* __restrict__ OH) {
  __shared__ alignas(16) bf16 Kts[64][64];  // [e][o_local]
  __shared__ alignas(16) bf16 Vts[64][64];  // [key_local][h]
  __shared__ float sc[16][64];
  __shared__ bf16  pl[16][64];
  __shared__ float row_m[16], row_l[16], row_corr[16];

  const int tid = threadIdx.x;
  const int lane = tid & 31, wave = tid >> 5;  // 4 waves
  const int q0 = blockIdx.x * 16;
  const int n  = blockIdx.y;
  const int b  = blockIdx.z;

  const bf16* Pbn = P + (size_t)b * S_ * D_ + (size_t)n * HD;   // + s*D_ per row
  const bf16* Tbn = T + ((size_t)b * NH + n) * (size_t)HD * O_; // + e*O_ per row

  if (tid < 16) { row_m[tid] = -3.4e38f; row_l[tid] = 0.f; }

  // Q fragments (head dim 64 = two K-steps of 32), loaded once from global
  v16bf qf0, qf1;
  {
    const int m = lane & 15;
    const int kb = (lane < 16) ? 0 : 16;
    const bf16* qp = Pbn + (size_t)(q0 + m) * D_;
#pragma unroll
    for (int v = 0; v < 8; ++v) {
      const int kk = kb + (v << 1);
      qf0[2 * v]     = qp[kk];
      qf0[2 * v + 1] = qp[kk + 1];
      qf1[2 * v]     = qp[32 + kk];
      qf1[2 * v + 1] = qp[32 + kk + 1];
    }
  }
  v8f oacc = {};
  const float inv_scale = 1.0f / (HD * 32.0f);  // hd*sqrt(S) = 2048

  for (int ko = 0; ko < O_; ko += 64) {
    __syncthreads();  // previous Kts/Vts/pl/row_corr fully consumed
    load_tile(&Kts[0][0], Tbn + ko, O_, 64, 64, tid, 128, wave);
    load_tile(&Vts[0][0], Pbn + (size_t)ko * D_, D_, 64, 64, tid, 128, wave);
    tiles_commit(wave);

    // --- score tile: 16x16 per wave over 64-deep head dim (2 WMMAs) ---
    v8f s = {};
    s = wmma_bf16(qf0, lds_bK64(Kts, lane, 0, wave * 16), s);
    s = wmma_bf16(qf1, lds_bK64(Kts, lane, 32, wave * 16), s);
    {
      const int cc = lane & 15;
      const int rb = (lane < 16) ? 0 : 8;
#pragma unroll
      for (int r = 0; r < 8; ++r) sc[rb + r][wave * 16 + cc] = s[r] * inv_scale;
    }
    __syncthreads();

    // --- online softmax over the 64-wide key tile ---
    {
      const int row = tid >> 3, seg = tid & 7;  // 16 rows x 8 segs x 8 cols
      float mloc = -3.4e38f;
#pragma unroll
      for (int j = 0; j < 8; ++j) mloc = fmaxf(mloc, sc[row][seg * 8 + j]);
#pragma unroll
      for (int off = 4; off >= 1; off >>= 1)
        mloc = fmaxf(mloc, __shfl_xor(mloc, off, 8));
      const float mold = row_m[row];
      const float mnew = fmaxf(mold, mloc);
      float ssum = 0.f;
#pragma unroll
      for (int j = 0; j < 8; ++j) {
        const float ev = __expf(sc[row][seg * 8 + j] - mnew);
        pl[row][seg * 8 + j] = (bf16)ev;
        ssum += ev;
      }
#pragma unroll
      for (int off = 4; off >= 1; off >>= 1) ssum += __shfl_xor(ssum, off, 8);
      if (seg == 0) {
        const float c = __expf(mold - mnew);
        row_corr[row] = c;
        row_l[row] = row_l[row] * c + ssum;
        row_m[row] = mnew;
      }
    }
    __syncthreads();

    // --- rescale accumulator, add P_tile @ V_tile (2 WMMAs) ---
    {
      const int rb = (lane < 16) ? 0 : 8;
#pragma unroll
      for (int r = 0; r < 8; ++r) oacc[r] *= row_corr[rb + r];
      oacc = wmma_bf16(lds_a64(pl, lane, 0, 0),
                       lds_bK64(Vts, lane, 0, wave * 16), oacc);
      oacc = wmma_bf16(lds_a64(pl, lane, 0, 32),
                       lds_bK64(Vts, lane, 32, wave * 16), oacc);
    }
  }
  __syncthreads();

  // finalize: divide by row sums, store bf16 head output in [B,S,D] layout
  {
    const int cc = lane & 15;
    const int rb = (lane < 16) ? 0 : 8;
#pragma unroll
    for (int r = 0; r < 8; ++r) {
      const float ol = oacc[r] / row_l[rb + r];
      OH[((size_t)b * S_ + q0 + rb + r) * D_ + (size_t)n * HD + wave * 16 + cc] =
          (bf16)ol;
    }
  }
}

// ---------------------------------------------------------------------------
// K5: out = A @ W^T + bias  (f32 out), same TDM-staged tiling as K1
// ---------------------------------------------------------------------------
__global__ void k_gemm_bias(const bf16* __restrict__ A, const bf16* __restrict__ W,
                            const float* __restrict__ bias, float* __restrict__ C,
                            int M, int K, int N) {
  __shared__ alignas(16) bf16 As[32][64];
  __shared__ alignas(16) bf16 Bs[64][64];
  const int tid  = threadIdx.x;
  const int lane = tid & 31, wave = tid >> 5;
  const int wm = wave >> 2, wn = wave & 3;
  const size_t rowblk = (size_t)blockIdx.x * 32;
  const size_t colblk = (size_t)blockIdx.y * 64;

  v8f acc = {};
  for (int k0 = 0; k0 < K; k0 += 64) {
    __syncthreads();
    load_tile(&As[0][0], A + rowblk * (size_t)K + k0, K, 32, 64, tid, 256, wave);
    load_tile(&Bs[0][0], W + colblk * (size_t)K + k0, K, 64, 64, tid, 256, wave);
    tiles_commit(wave);
    v16bf a = lds_a64(As, lane, wm * 16, 0);
    v16bf b = lds_bT64(Bs, lane, wn * 16, 0);
    acc = wmma_bf16(a, b, acc);
    a = lds_a64(As, lane, wm * 16, 32);
    b = lds_bT64(Bs, lane, wn * 16, 32);
    acc = wmma_bf16(a, b, acc);
  }
  const int cc = lane & 15;
  const int rb = (lane < 16) ? 0 : 8;
  const float bb = bias[colblk + wn * 16 + cc];
#pragma unroll
  for (int r = 0; r < 8; ++r)
    C[(rowblk + wm * 16 + rb + r) * (size_t)N + colblk + wn * 16 + cc] =
        acc[r] + bb;
}

// ---------------------------------------------------------------------------
// Host-side launch
// ---------------------------------------------------------------------------
extern "C" void kernel_launch(void* const* d_in, const int* in_sizes, int n_in,
                              void* d_out, int out_size, void* d_ws, size_t ws_size,
                              hipStream_t stream) {
  (void)in_sizes; (void)n_in; (void)out_size; (void)ws_size;
  const float* x      = (const float*)d_in[0];
  const float* attn_w = (const float*)d_in[1];
  const float* w_in   = (const float*)d_in[2];
  const float* b_in   = (const float*)d_in[3];
  const float* w_out  = (const float*)d_in[4];
  const float* b_out  = (const float*)d_in[5];
  float* out = (float*)d_out;

  // workspace carve-out (~36 MiB total)
  char* ws = (char*)d_ws;
  size_t off = 0;
  auto carve = [&](size_t bytes) -> void* {
    void* p = ws + off;
    off += (bytes + 255) & ~(size_t)255;
    return p;
  };
  bf16*  x_bf    = (bf16*)carve((size_t)B_ * S_ * D_ * 2);
  bf16*  win_bf  = (bf16*)carve((size_t)D_ * D_ * 2);
  bf16*  wout_bf = (bf16*)carve((size_t)D_ * D_ * 2);
  bf16*  p_bf    = (bf16*)carve((size_t)B_ * S_ * D_ * 2);
  float* xsum    = (float*)carve((size_t)B_ * NH * HD * 4);
  bf16*  t_bf    = (bf16*)carve((size_t)B_ * NH * HD * O_ * 2);
  bf16*  oh_bf   = (bf16*)carve((size_t)B_ * S_ * D_ * 2);

  const int nx = B_ * S_ * D_;
  const int nw = D_ * D_;
  k_cvt<<<dim3((nx + 255) / 256), dim3(256), 0, stream>>>(x, x_bf, nx);
  k_cvt<<<dim3((nw + 255) / 256), dim3(256), 0, stream>>>(w_in, win_bf, nw);
  k_cvt<<<dim3((nw + 255) / 256), dim3(256), 0, stream>>>(w_out, wout_bf, nw);

  k_gemm_softmax<<<dim3((B_ * S_) / 32, D_ / 64), dim3(256), 0, stream>>>(
      x_bf, win_bf, b_in, p_bf, B_ * S_, D_, D_);

  k_xsum<<<dim3(B_ * NH), dim3(256), 0, stream>>>(p_bf, xsum);

  k_contract<<<dim3(O_ / 8, HD, NH), dim3(256), 0, stream>>>(attn_w, xsum, t_bf);

  k_flash<<<dim3(S_ / 16, NH, B_), dim3(128), 0, stream>>>(p_bf, t_bf, oh_bf);

  k_gemm_bias<<<dim3((B_ * S_) / 32, D_ / 64), dim3(256), 0, stream>>>(
      oh_bf, wout_bf, b_out, out, B_ * S_, D_, D_);
}
